// DynamicGEFM_30863634989614
// MI455X (gfx1250) — matrix operand order repaired and checked by reading, not compiled, not run on hardware
//
#include <hip/hip_runtime.h>
#include <hip/hip_bf16.h>
#include <math.h>

// ---------------------------------------------------------------------------
// DynamicGEFM on MI455X (gfx1250, wave32):
//   pass 1: 3x conv3x3+BN+ReLU (f32 VALU) -> f16 Q [B,N,C] and K [B,C,N]
//   pass 2: fused flash-attention (v_wmma_f32_16x16x32_f16) over both
//           modalities, online softmax (DPP row_xmask reductions),
//           gamma*(rgb+depth) combine.
// Shapes: B=4, C=64, H=W=64, N=4096.
// Workspace: Qh 2MB @0, Kr 2MB @2MB, Kd 2MB @4MB (6MB total).
// ---------------------------------------------------------------------------

typedef __attribute__((ext_vector_type(16))) _Float16 v16h;
typedef __attribute__((ext_vector_type(8)))  float    v8f;

union V16HU { v16h v; uint4 q[2]; };

#define EPS_BN 1e-5f

// --- cross-lane reductions within 16-lane row groups --------------------
// DPP ROW_XMASK (0x160|mask): lane <- lane^mask within each 16-lane row.
// Pure VALU (no ds_bpermute). masks 1,2,4,8 never cross the 16-lane group.
#if defined(__has_builtin)
#if __has_builtin(__builtin_amdgcn_update_dpp)
#define GEFM_USE_DPP 1
#endif
#endif

#ifdef GEFM_USE_DPP
#define DPP_XOR(v, m)                                                     \
    __int_as_float(__builtin_amdgcn_update_dpp(                           \
        __float_as_int(v), __float_as_int(v), 0x160 + (m), 0xF, 0xF, true))
#else
#define DPP_XOR(v, m) __shfl_xor((v), (m))
#endif

__device__ __forceinline__ float row_max16(float v) {
    v = fmaxf(v, DPP_XOR(v, 1));
    v = fmaxf(v, DPP_XOR(v, 2));
    v = fmaxf(v, DPP_XOR(v, 4));
    v = fmaxf(v, DPP_XOR(v, 8));
    return v;
}
__device__ __forceinline__ float row_sum16(float v) {
    v += DPP_XOR(v, 1);
    v += DPP_XOR(v, 2);
    v += DPP_XOR(v, 4);
    v += DPP_XOR(v, 8);
    return v;
}

// ---------------------------------------------------------------------------
// Conv 3x3 pad=1 + BN(inference) + ReLU, f32 in, f16 out.
// x1 supplies channels 64..127 when Cin==128 (concat(rgb,depth)).
// n_major=1 -> out[b][n][co] (Q layout), else out[b][co][n] (K layout).
// ---------------------------------------------------------------------------
__global__ void __launch_bounds__(256)
conv_bn_relu_kernel(const float* __restrict__ x0,
                    const float* __restrict__ x1,
                    const float* __restrict__ w,   // [64][Cin][3][3]
                    const float* __restrict__ gs,
                    const float* __restrict__ bs,
                    const float* __restrict__ ms,
                    const float* __restrict__ vs,
                    _Float16* __restrict__ out,
                    int Cin, int n_major)
{
    const int N = 4096;
    int idx = blockIdx.x * blockDim.x + threadIdx.x;   // b*64*4096 total
    int n  = idx & (N - 1);
    int co = (idx >> 12) & 63;
    int b  = idx >> 18;
    int h  = n >> 6;
    int x  = n & 63;

    float acc = 0.f;
    const float* wco = w + (size_t)co * Cin * 9;
    for (int ci = 0; ci < Cin; ++ci) {
        const float* xb = (ci < 64)
            ? (x0 + ((size_t)(b * 64 + ci) << 12))
            : (x1 + ((size_t)(b * 64 + (ci - 64)) << 12));
        const float* wp = wco + ci * 9;
#pragma unroll
        for (int dh = -1; dh <= 1; ++dh) {
            int hh = h + dh;
            if ((unsigned)hh >= 64u) continue;
#pragma unroll
            for (int dw = -1; dw <= 1; ++dw) {
                int ww = x + dw;
                if ((unsigned)ww >= 64u) continue;
                acc = fmaf(xb[(hh << 6) + ww], wp[(dh + 1) * 3 + (dw + 1)], acc);
            }
        }
    }
    float sc = gs[co] * rsqrtf(vs[co] + EPS_BN);
    float sh = bs[co] - ms[co] * sc;
    float y  = fmaxf(fmaf(acc, sc, sh), 0.f);
    if (n_major) out[((size_t)(b * N + n) << 6) + co]  = (_Float16)y;
    else         out[((size_t)(b * 64 + co) << 12) + n] = (_Float16)y;
}

// ---------------------------------------------------------------------------
// Fused dual-modality flash attention.
// Grid: 128 blocks (4 batches x 32 row-blocks of 128), 256 threads = 8 waves.
// Each wave owns a 16-row (n) tile; K streamed in 128-column LDS tiles,
// softmax state updated per 64-column group.
// ---------------------------------------------------------------------------
__global__ void __launch_bounds__(256, 1)
fused_attention_kernel(const _Float16* __restrict__ Qh,   // [B][N][64]
                       const _Float16* __restrict__ Kr,   // [B][64][N]
                       const _Float16* __restrict__ Kd,   // [B][64][N]
                       const float* __restrict__ gamma,
                       float* __restrict__ out)           // [B][64][N]
{
    const int N = 4096, C = 64;

    // Padded strides (72/136 halves) -> conflict-free ds_load_b128 operands.
    __shared__ _Float16 lds_mc[128][72];     // K tile [m][c]      18.0 KB
    __shared__ _Float16 lds_cm[64][136];     // K tile [c][m]      17.0 KB
    __shared__ _Float16 lds_p[8][16][72];    // per-wave P (64 col) 18.0 KB

    const int tid  = threadIdx.x;
    const int wave = tid >> 5;
    const int lane = tid & 31;
    const int b    = blockIdx.x >> 5;
    const int rb   = blockIdx.x & 31;
    const int n0   = rb * 128 + wave * 16;
    const int row  = lane & 15;              // A rows / B cols / C-D column
    const int hi   = lane >> 4;

    // --- wave's Q A-operand tiles (16 rows x 64 c, two k=32 steps).
    // A layout (ISA 7.12.2): lane holds row (lane&15); halves are two
    // contiguous 8-element runs at k = hi*8 and 16+hi*8 (+32 for step 1).
    V16HU aQ[2];
    {
        const _Float16* qrow = Qh + ((size_t)(b * N + n0 + row) << 6);
#pragma unroll
        for (int s = 0; s < 2; ++s) {
            aQ[s].q[0] = *(const uint4*)(qrow + s * 32 + hi * 8);
            aQ[s].q[1] = *(const uint4*)(qrow + s * 32 + 16 + hi * 8);
        }
    }

    float osum[4][8];
#pragma unroll
    for (int cb = 0; cb < 4; ++cb)
#pragma unroll
        for (int r = 0; r < 8; ++r) osum[cb][r] = 0.f;

    for (int mod = 0; mod < 2; ++mod) {
        const _Float16* Kbase = (mod ? Kd : Kr) + ((size_t)b * C * N);

        v8f acc[4];
#pragma unroll
        for (int cb = 0; cb < 4; ++cb) acc[cb] = (v8f){0,0,0,0,0,0,0,0};
        float rmax[8], rsum[8];
#pragma unroll
        for (int r = 0; r < 8; ++r) { rmax[r] = -__builtin_inff(); rsum[r] = 0.f; }

        for (int t = 0; t < 32; ++t) {              // 32 tiles of 128 m-cols
            const int m0 = t << 7;
            __syncthreads();                        // WAR on LDS K tile
            {   // cooperative K[:, m0:m0+128]: 2 channels x 16 cols / thread,
                // dual layout; channel pairs pack transposed stores into b32.
                int c0   = (tid >> 3) << 1;          // 0,2,..,62
                int mseg = (tid & 7) << 4;           // 0,16,..,112
                const _Float16* s0 = Kbase + (size_t)c0 * N + m0 + mseg;
                const _Float16* s1 = s0 + N;
                union { uint4 q[2]; unsigned short us[16]; } r0, r1;
                r0.q[0] = *(const uint4*)(s0);
                r0.q[1] = *(const uint4*)(s0 + 8);
                r1.q[0] = *(const uint4*)(s1);
                r1.q[1] = *(const uint4*)(s1 + 8);
                if (t < 31) {                        // next tile -> L2/WGP$
                    __builtin_prefetch((const void*)(s0 + 128), 0, 1);
                    __builtin_prefetch((const void*)(s1 + 128), 0, 1);
                }
                *(uint4*)&lds_cm[c0][mseg]         = r0.q[0];
                *(uint4*)&lds_cm[c0][mseg + 8]     = r0.q[1];
                *(uint4*)&lds_cm[c0 + 1][mseg]     = r1.q[0];
                *(uint4*)&lds_cm[c0 + 1][mseg + 8] = r1.q[1];
#pragma unroll
                for (int j = 0; j < 16; ++j) {       // packed (c0,c0+1) pair
                    unsigned pk = (unsigned)r0.us[j] |
                                  ((unsigned)r1.us[j] << 16);
                    *(unsigned*)&lds_mc[mseg + j][c0] = pk;
                }
            }
            __syncthreads();

            for (int hg = 0; hg < 2; ++hg) {        // 64-col groups
                const int g0 = hg << 6;

                // ---- logits L[16n x 64m] = Q(16x64) . K(64 x 64m) ----
                v8f L[4];
#pragma unroll
                for (int mh = 0; mh < 4; ++mh) {
                    const int mloc = g0 + mh * 16 + row;    // B col = lane&15
                    V16HU b1a, b1b;                          // k = hi*8 + e, 16 + hi*8 + e
                    b1a.q[0] = *(const uint4*)&lds_mc[mloc][hi * 8];
                    b1a.q[1] = *(const uint4*)&lds_mc[mloc][16 + hi * 8];
                    b1b.q[0] = *(const uint4*)&lds_mc[mloc][32 + hi * 8];
                    b1b.q[1] = *(const uint4*)&lds_mc[mloc][48 + hi * 8];
                    v8f z = (v8f){0,0,0,0,0,0,0,0};
                    z = __builtin_amdgcn_wmma_f32_16x16x32_f16(
                            false, aQ[0].v, false, b1a.v, (short)0, z, false, false);
                    z = __builtin_amdgcn_wmma_f32_16x16x32_f16(
                            false, aQ[1].v, false, b1b.v, (short)0, z, false, false);
                    L[mh] = z;
                }

                // ---- online softmax over the 64 group columns (DPP) ----
#pragma unroll
                for (int r = 0; r < 8; ++r) {
                    float mx = fmaxf(fmaxf(L[0][r], L[1][r]),
                                     fmaxf(L[2][r], L[3][r]));
                    mx = row_max16(mx);
                    float nm = fmaxf(rmax[r], mx);
                    float sc = __expf(rmax[r] - nm);  // exp(-inf)=0 first hit
                    rmax[r] = nm;
                    float p0 = __expf(L[0][r] - nm);
                    float p1 = __expf(L[1][r] - nm);
                    float p2 = __expf(L[2][r] - nm);
                    float p3 = __expf(L[3][r] - nm);
                    float s = row_sum16((p0 + p1) + (p2 + p3));
                    rsum[r] = rsum[r] * sc + s;
#pragma unroll
                    for (int cb = 0; cb < 4; ++cb) acc[cb][r] *= sc;
                    const int rr = r + hi * 8;        // C/D: M = r + hi*8
                    lds_p[wave][rr][row]      = (_Float16)p0;   // col lane&15
                    lds_p[wave][rr][16 + row] = (_Float16)p1;
                    lds_p[wave][rr][32 + row] = (_Float16)p2;
                    lds_p[wave][rr][48 + row] = (_Float16)p3;
                }
                // same-wave DS ops are in-order; fence compiler + counter
                __asm__ volatile("s_wait_dscnt 0x0" ::: "memory");

                // ---- out[16n x 64c] += P(16 x 64m) . K^T(64m x 64c) ----
#pragma unroll
                for (int ch = 0; ch < 2; ++ch) {      // 32-m chunks of group
                    const int mc0 = ch << 5;
                    V16HU a2;                          // A row = lane&15
                    a2.q[0] = *(const uint4*)&lds_p[wave][row][mc0 + hi * 8];
                    a2.q[1] = *(const uint4*)&lds_p[wave][row][mc0 + 16 + hi * 8];
#pragma unroll
                    for (int cb = 0; cb < 4; ++cb) {
                        V16HU b2;                      // k = m = hi*8 + e, 16 + hi*8 + e
                        const _Float16* p =
                            &lds_cm[cb * 16 + row][g0 + mc0 + hi * 8];
                        b2.q[0] = *(const uint4*)p;
                        b2.q[1] = *(const uint4*)(p + 16);
                        acc[cb] = __builtin_amdgcn_wmma_f32_16x16x32_f16(
                            false, a2.v, false, b2.v, (short)0, acc[cb],
                            false, false);
                    }
                }
            } // 64-col group
        } // tile

#pragma unroll
        for (int r = 0; r < 8; ++r) {
            float inv = 1.0f / rsum[r];
#pragma unroll
            for (int cb = 0; cb < 4; ++cb) osum[cb][r] += acc[cb][r] * inv;
        }
    } // modality

    const float gv = gamma[0];
#pragma unroll
    for (int cb = 0; cb < 4; ++cb) {
#pragma unroll
        for (int r = 0; r < 8; ++r) {
            int c  = cb * 16 + row;                      // C/D col = lane&15
            int nn = n0 + r + hi * 8;                    // C/D row = r + hi*8
            out[(((size_t)b * 64 + c) << 12) + nn] = gv * osum[cb][r];
        }
    }
}

// ---------------------------------------------------------------------------
extern "C" void kernel_launch(void* const* d_in, const int* in_sizes, int n_in,
                              void* d_out, int out_size, void* d_ws, size_t ws_size,
                              hipStream_t stream)
{
    (void)in_sizes; (void)n_in; (void)out_size; (void)ws_size;

    const float* rgb   = (const float*)d_in[0];
    const float* depth = (const float*)d_in[1];
    const float* Wq    = (const float*)d_in[2];
    const float* Wr    = (const float*)d_in[3];
    const float* Wd    = (const float*)d_in[4];
    const float* g_q = (const float*)d_in[5],  *b_q = (const float*)d_in[6],
               * m_q = (const float*)d_in[7],  *v_q = (const float*)d_in[8];
    const float* g_r = (const float*)d_in[9],  *b_r = (const float*)d_in[10],
               * m_r = (const float*)d_in[11], *v_r = (const float*)d_in[12];
    const float* g_d = (const float*)d_in[13], *b_d = (const float*)d_in[14],
               * m_d = (const float*)d_in[15], *v_d = (const float*)d_in[16];
    const float* gamma = (const float*)d_in[17];

    char* ws = (char*)d_ws;
    _Float16* Qh = (_Float16*)(ws);                       // 2 MB [B][N][64]
    _Float16* Kr = (_Float16*)(ws + (size_t)(2u << 20));  // 2 MB [B][64][N]
    _Float16* Kd = (_Float16*)(ws + (size_t)(4u << 20));  // 2 MB [B][64][N]

    // pass 1: convs (1M outputs each, 256 threads/block)
    conv_bn_relu_kernel<<<4096, 256, 0, stream>>>(
        rgb, depth, Wq, g_q, b_q, m_q, v_q, Qh, 128, 1);
    conv_bn_relu_kernel<<<4096, 256, 0, stream>>>(
        rgb, rgb,   Wr, g_r, b_r, m_r, v_r, Kr, 64, 0);
    conv_bn_relu_kernel<<<4096, 256, 0, stream>>>(
        depth, depth, Wd, g_d, b_d, m_d, v_d, Kd, 64, 0);

    // pass 2: fused dual attention + combine
    fused_attention_kernel<<<128, 256, 0, stream>>>(
        Qh, Kr, Kd, gamma, (float*)d_out);
}
